// BDLOTreeLSTM_42984032698734
// MI455X (gfx1250) — matrix operand order repaired and checked by reading, not compiled, not run on hardware
//
#include <hip/hip_runtime.h>
#include <math.h>

// ---------------------------------------------------------------------------
// BDLO Tree-LSTM forward for MI455X (gfx1250, wave32, WMMA).
// f16 operands / f32 accumulate via v_wmma_f32_16x16x32_f16.
// All GEMMs are LDS-free: A fragments read directly from row-major f16 global
// (2x b128 per lane), B fragments from weights pre-swizzled into a contiguous-k (k = 16*(lane/16) + j)
// wave32 WMMA B layout (2x b128 per lane, fully coalesced, L2-resident).
// ---------------------------------------------------------------------------

typedef _Float16 f16;
typedef __attribute__((ext_vector_type(16))) _Float16 v16h;
typedef __attribute__((ext_vector_type(8)))  _Float16 v8h;
typedef __attribute__((ext_vector_type(8)))  float    v8f;

#define BSZ 512
#define NBR 3
#define NVT 64
#define HD  256
#define CLEN 40
#define NTOK (NBR * NVT)          // 192
#define NROW (NTOK * BSZ)         // 98304
#define YLD  1280                 // step y stride: 768 iou | 256 f | 256 fchild

__device__ __forceinline__ float sigf(float x) { return 1.0f / (1.0f + __expf(-x)); }

// ---- fragment helpers (wave32, v_wmma_f32_16x16x32_f16) --------------------

// A 16x32 from row-major f16 global: lane<16 -> M=lane, K {0..15};
// lane>=16 -> M=lane-16, K {16..31}. Two 16B loads, shuffle-merged.
__device__ __forceinline__ v16h load_frag_a(const f16* __restrict__ tile, int ld) {
  int lane = threadIdx.x & 31;
  int m  = lane & 15;
  int kb = (lane >> 4) << 4;                 // 0 or 16
  const f16* p = tile + (size_t)m * ld + kb;
  v8h lo = *(const v8h*)p;                   // K = kb .. kb+7
  v8h hi = *(const v8h*)(p + 8);             // K = kb+8 .. kb+15
  return __builtin_shufflevector(lo, hi, 0, 1, 2, 3, 4, 5, 6, 7,
                                         8, 9, 10, 11, 12, 13, 14, 15);
}

// B 32x16 from a pre-swizzled 512-half blob: lane reads halves [lane*16, +16).
__device__ __forceinline__ v16h load_frag_b(const f16* __restrict__ blob) {
  int lane = threadIdx.x & 31;
  const f16* p = blob + lane * 16;
  v8h lo = *(const v8h*)p;
  v8h hi = *(const v8h*)(p + 8);
  return __builtin_shufflevector(lo, hi, 0, 1, 2, 3, 4, 5, 6, 7,
                                         8, 9, 10, 11, 12, 13, 14, 15);
}

// C/D 16x16 f32: VGPR i -> (M=i, N=lane) lanes 0-15, (M=i+8, N=lane-16) 16-31
__device__ __forceinline__ void store_c_f32(const v8f& acc, float* out, int ldc) {
  int lane = threadIdx.x & 31;
  int n  = lane & 15;
  int mh = (lane >> 4) << 3;
#pragma unroll
  for (int i = 0; i < 8; ++i) out[(mh + i) * ldc + n] = acc[i];
}

__device__ __forceinline__ void store_c_f16(const v8f& acc, const float* bias,
                                            int relu, f16* out, int ldc) {
  int lane = threadIdx.x & 31;
  int n  = lane & 15;
  int mh = (lane >> 4) << 3;
#pragma unroll
  for (int i = 0; i < 8; ++i) {
    float v = acc[i] + bias[n];
    if (relu) v = fmaxf(v, 0.0f);
    out[(mh + i) * ldc + n] = (f16)v;
  }
}

__device__ __forceinline__ v8f zero_acc() {
  v8f a;
#pragma unroll
  for (int i = 0; i < 8; ++i) a[i] = 0.0f;
  return a;
}

#define WMMA(A, Bf, C) __builtin_amdgcn_wmma_f32_16x16x32_f16(false, (A), false, (Bf), (short)0, (C), false, false)

// ---------------------------------------------------------------------------
// Weight conversion + swizzle into the contiguous-k WMMA B layout read by load_frag_b.
// src f32 [K][N] row-major -> blobs of 512 halves per (16-col x 32-K) block:
// dst[ (n16*(K/32) + k32)*512 + lane*16 + j ],  lane = (kk>=16)*16 + (n&15),
// j = kk & 15   (kk = k & 31)
// ---------------------------------------------------------------------------
__global__ void swz_kernel(const float* __restrict__ src, f16* __restrict__ dst,
                           int K, int N) {
  int i = blockIdx.x * blockDim.x + threadIdx.x;
  if (i >= K * N) return;
  int k = i / N, n = i % N;
  int kk   = k & 31;
  int lane = ((kk >> 4) << 4) | (n & 15);
  int j    = k & 15;
  size_t blk = (size_t)(n >> 4) * (K >> 5) + (k >> 5);
  dst[blk * 512 + lane * 16 + j] = (f16)src[i];
}

// ---------------------------------------------------------------------------
// Encoder layer 1: feats(9) @ enc_w1 + b1, ReLU -> f16 [NROW][256]
// ---------------------------------------------------------------------------
__global__ __launch_bounds__(256)
void enc1_kernel(const float* __restrict__ cv, const float* __restrict__ pv,
                 const float* __restrict__ hints,
                 const float* __restrict__ w1, const float* __restrict__ b1,
                 f16* __restrict__ out) {
  int r   = blockIdx.x;
  int col = threadIdx.x;
  int t = r / BSZ, b = r % BSZ;
  int nb = t / NVT, v = t % NVT;
  int base = ((b * NBR + nb) * NVT + v) * 3;
  float f[9];
#pragma unroll
  for (int d = 0; d < 3; ++d) {
    float c = cv[base + d];
    f[d]     = c;
    f[3 + d] = c - pv[base + d];
    f[6 + d] = hints[base + d];
  }
  float acc = b1[col];
#pragma unroll
  for (int i = 0; i < 9; ++i) acc += f[i] * w1[i * HD + col];
  out[(size_t)r * HD + col] = (f16)fmaxf(acc, 0.0f);
}

// ---------------------------------------------------------------------------
// Generic LDS-free WMMA GEMM: A f16 [M][K] row-major, B swizzled, bias/relu.
// Block 128 thr = 4 independent waves; wave handles 16 rows x 64 cols.
// grid: (N/64, M/64)
// ---------------------------------------------------------------------------
__global__ __launch_bounds__(128)
void gemm_af16_kernel(const f16* __restrict__ A, const f16* __restrict__ Bsw,
                      const float* __restrict__ bias, f16* __restrict__ out,
                      int N, int K, int relu) {
  int wave = threadIdx.x >> 5;
  int row0 = blockIdx.y * 64 + wave * 16;
  int nb16 = blockIdx.x * 4;
  int K32  = K >> 5;
  v8f acc[4];
#pragma unroll
  for (int t = 0; t < 4; ++t) acc[t] = zero_acc();

  for (int k0 = 0; k0 < K; k0 += 32) {
    v16h a = load_frag_a(A + (size_t)row0 * K + k0, K);
#pragma unroll
    for (int t = 0; t < 4; ++t) {
      v16h b = load_frag_b(Bsw + ((size_t)(nb16 + t) * K32 + (k0 >> 5)) * 512);
      acc[t] = WMMA(a, b, acc[t]);
    }
  }
#pragma unroll
  for (int t = 0; t < 4; ++t) {
    f16* o = out + (size_t)row0 * N + nb16 * 16 + t * 16;
    store_c_f16(acc[t], bias + nb16 * 16 + t * 16, relu, o, N);
  }
}

// ---------------------------------------------------------------------------
// Decoder layer 1: A = concat(buHh, tdHh) f16, K=512, B = dec_w1 swizzled.
// ---------------------------------------------------------------------------
__global__ __launch_bounds__(128)
void gemm_dec_kernel(const f16* __restrict__ A0, const f16* __restrict__ A1,
                     const f16* __restrict__ Bsw, const float* __restrict__ bias,
                     f16* __restrict__ out) {
  int wave = threadIdx.x >> 5;
  int row0 = blockIdx.y * 64 + wave * 16;
  int nb16 = blockIdx.x * 4;
  const int K32 = 512 >> 5;   // 16 K-blocks in swizzled dec_w1
  v8f acc[4];
#pragma unroll
  for (int t = 0; t < 4; ++t) acc[t] = zero_acc();

  for (int k0 = 0; k0 < 512; k0 += 32) {
    const f16* src = (k0 < 256) ? A0 : A1;
    int kloc = (k0 < 256) ? k0 : (k0 - 256);
    v16h a = load_frag_a(src + (size_t)row0 * HD + kloc, HD);
#pragma unroll
    for (int t = 0; t < 4; ++t) {
      v16h b = load_frag_b(Bsw + ((size_t)(nb16 + t) * K32 + (k0 >> 5)) * 512);
      acc[t] = WMMA(a, b, acc[t]);
    }
  }
#pragma unroll
  for (int t = 0; t < 4; ++t) {
    f16* o = out + (size_t)row0 * HD + nb16 * 16 + t * 16;
    store_c_f16(acc[t], bias + nb16 * 16 + t * 16, /*relu=*/1, o, HD);
  }
}

// ---------------------------------------------------------------------------
// Per-step GEMM: y = x@W + h@U per column group. All operands f16.
//   mode 0 (BU):  cb<12  iou (y 0..767):    h = hPrev (+ chH)  vs U_iou
//                 cb<16  f   (y 768..1023): h = hPrev          vs U_f
//                 cb<20  fch (y 1024..):    h = chH            vs U_f
//   mode 1 (TD):  16 blocks (y 0..1023): x@W_ih + hPrev@W_hh
// grid: (colBlocks, BSZ/64), block 128 (4 independent waves).
// ---------------------------------------------------------------------------
__global__ __launch_bounds__(128)
void step_gemm_kernel(const f16* __restrict__ x, const f16* __restrict__ hPrev,
                      const f16* __restrict__ chH,
                      const f16* __restrict__ Wsw, const f16* __restrict__ Usw,
                      const f16* __restrict__ WfSw, const f16* __restrict__ UfSw,
                      float* __restrict__ y, int mode) {
  int cb   = blockIdx.x;
  int wave = threadIdx.x >> 5;
  int row0 = blockIdx.y * 64 + wave * 16;
  const int K32 = HD >> 5;    // 8

  int nb16, ybase;
  const f16 *Wp, *Up, *hA;
  bool addCh = false;
  if (mode == 1) {                       // top-down: Wsw=W_ih, Usw=W_hh
    nb16 = cb * 4; ybase = 0; Wp = Wsw; Up = Usw; hA = hPrev;
  } else if (cb < 12) {                  // iou
    nb16 = cb * 4; ybase = 0; Wp = Wsw; Up = Usw; hA = hPrev;
    addCh = (chH != nullptr);
  } else if (cb < 16) {                  // f (self)
    nb16 = (cb - 12) * 4; ybase = 768; Wp = WfSw; Up = UfSw; hA = hPrev;
  } else {                               // f (child)
    nb16 = (cb - 16) * 4; ybase = 1024; Wp = WfSw; Up = UfSw; hA = chH;
  }
  bool hasH = (hA != nullptr);

  v8f acc[4];
#pragma unroll
  for (int t = 0; t < 4; ++t) acc[t] = zero_acc();

  for (int k0 = 0; k0 < HD; k0 += 32) {
    v16h ax = load_frag_a(x + (size_t)row0 * HD + k0, HD);
    v16h ah;
    if (hasH) {
      ah = load_frag_a(hA + (size_t)row0 * HD + k0, HD);
      if (addCh) ah = ah + load_frag_a(chH + (size_t)row0 * HD + k0, HD);
    }
#pragma unroll
    for (int t = 0; t < 4; ++t) {
      size_t wb = ((size_t)(nb16 + t) * K32 + (k0 >> 5)) * 512;
      v16h bw = load_frag_b(Wp + wb);
      acc[t] = WMMA(ax, bw, acc[t]);
      if (hasH) {
        v16h bu = load_frag_b(Up + wb);
        acc[t] = WMMA(ah, bu, acc[t]);
      }
    }
  }
#pragma unroll
  for (int t = 0; t < 4; ++t) {
    float* o = y + (size_t)row0 * YLD + ybase + nb16 * 16 + t * 16;
    store_c_f32(acc[t], o, YLD);
  }
}

// ---------------------------------------------------------------------------
// Bottom-up LSTM cell: writes c (f32) and h (f16 mirror for GEMM A-operand)
// ---------------------------------------------------------------------------
__global__ __launch_bounds__(256)
void bu_cell_kernel(const float* __restrict__ y,
                    const float* __restrict__ bIou, const float* __restrict__ bF,
                    const float* __restrict__ cPrev, const float* __restrict__ chC,
                    f16* __restrict__ outHh, float* __restrict__ outC) {
  int idx = blockIdx.x * blockDim.x + threadIdx.x;   // BSZ*HD
  int b = idx >> 8, j = idx & 255;
  const float* yr = y + (size_t)b * YLD;
  float iv = yr[j]       + bIou[j];
  float ov = yr[256 + j] + bIou[256 + j];
  float uv = yr[512 + j] + bIou[512 + j];
  float fv = yr[768 + j] + bF[j];
  float c = sigf(iv) * tanhf(uv) + sigf(fv) * (cPrev ? cPrev[idx] : 0.0f);
  if (chC) c += sigf(yr[1024 + j] + bF[j]) * chC[idx];
  float h = sigf(ov) * tanhf(c);
  outHh[idx] = (f16)h;
  outC[idx]  = c;
}

// ---------------------------------------------------------------------------
// Top-down LSTM cell
// ---------------------------------------------------------------------------
__global__ __launch_bounds__(256)
void td_cell_kernel(const float* __restrict__ y,
                    const float* __restrict__ bIh, const float* __restrict__ bHh,
                    const float* __restrict__ cPrev,
                    f16* __restrict__ outHh, float* __restrict__ outC) {
  int idx = blockIdx.x * blockDim.x + threadIdx.x;
  int b = idx >> 8, j = idx & 255;
  const float* yr = y + (size_t)b * YLD;
  float iv = yr[j]       + bIh[j]       + bHh[j];
  float fv = yr[256 + j] + bIh[256 + j] + bHh[256 + j];
  float gv = yr[512 + j] + bIh[512 + j] + bHh[512 + j];
  float ov = yr[768 + j] + bIh[768 + j] + bHh[768 + j];
  float c = sigf(fv) * cPrev[idx] + sigf(iv) * tanhf(gv);
  float h = sigf(ov) * tanhf(c);
  outHh[idx] = (f16)h;
  outC[idx]  = c;
}

// ---------------------------------------------------------------------------
// Seed copy for top-down root: h (f16) and c (f32) from bottom-up root
// ---------------------------------------------------------------------------
__global__ __launch_bounds__(256)
void seed_kernel(const f16* __restrict__ srcH, const float* __restrict__ srcC,
                 f16* __restrict__ dstH, float* __restrict__ dstC) {
  int idx = blockIdx.x * blockDim.x + threadIdx.x;
  dstH[idx] = srcH[idx];
  dstC[idx] = srcC[idx];
}

// ---------------------------------------------------------------------------
// Decoder layer 2 + residual + mask
// ---------------------------------------------------------------------------
__global__ __launch_bounds__(256)
void dec_out_kernel(const f16* __restrict__ hid, const float* __restrict__ w2,
                    const float* __restrict__ b2, const float* __restrict__ cv,
                    float* __restrict__ pred) {
  int r = blockIdx.x * blockDim.x + threadIdx.x;
  if (r >= NROW) return;
  int t = r / BSZ, b = r % BSZ;
  int nb = t / NVT, v = t % NVT;
  bool masked = (nb > 0) && (v >= CLEN);
  const f16* h = hid + (size_t)r * HD;
  float a0 = b2[0], a1 = b2[1], a2 = b2[2];
  for (int k = 0; k < HD; ++k) {
    float hv = (float)h[k];
    a0 += hv * w2[k * 3 + 0];
    a1 += hv * w2[k * 3 + 1];
    a2 += hv * w2[k * 3 + 2];
  }
  int ob = ((b * NBR + nb) * NVT + v) * 3;
  pred[ob + 0] = masked ? 0.0f : (cv[ob + 0] + a0);
  pred[ob + 1] = masked ? 0.0f : (cv[ob + 1] + a1);
  pred[ob + 2] = masked ? 0.0f : (cv[ob + 2] + a2);
}

// ---------------------------------------------------------------------------
// Host orchestration
// ---------------------------------------------------------------------------
extern "C" void kernel_launch(void* const* d_in, const int* in_sizes, int n_in,
                              void* d_out, int out_size, void* d_ws, size_t ws_size,
                              hipStream_t stream) {
  const float* cv     = (const float*)d_in[0];
  const float* pv     = (const float*)d_in[1];
  const float* hints  = (const float*)d_in[2];
  const float* enc_w1 = (const float*)d_in[3];
  const float* enc_b1 = (const float*)d_in[4];
  const float* enc_w2 = (const float*)d_in[5];
  const float* enc_b2 = (const float*)d_in[6];
  const float* W_iou  = (const float*)d_in[7];
  const float* b_iou  = (const float*)d_in[8];
  const float* U_iou  = (const float*)d_in[9];
  const float* W_f    = (const float*)d_in[10];
  const float* b_f    = (const float*)d_in[11];
  const float* U_f    = (const float*)d_in[12];
  const float* W_ih   = (const float*)d_in[13];
  const float* b_ih   = (const float*)d_in[14];
  const float* W_hh   = (const float*)d_in[15];
  const float* b_hh   = (const float*)d_in[16];
  const float* dec_w1 = (const float*)d_in[17];
  const float* dec_b1 = (const float*)d_in[18];
  const float* dec_w2 = (const float*)d_in[19];
  const float* dec_b2 = (const float*)d_in[20];
  float* pred = (float*)d_out;

  // workspace carve-out (~340 MB required)
  char* base = (char*)d_ws;
  size_t off = 0;
  auto alloc = [&](size_t bytes) -> void* {
    void* r = base + off;
    off += (bytes + 255) & ~(size_t)255;
    return r;
  };
  const size_t BH  = (size_t)BSZ * HD;          // 131072
  const size_t ALL = (size_t)NROW * HD;         // 25.2M
  f16*   encH   = (f16*)  alloc(ALL * sizeof(f16));
  f16*   tmpH   = (f16*)  alloc(ALL * sizeof(f16));   // enc1 out, then dec hidden
  f16*   buHh   = (f16*)  alloc(ALL * sizeof(f16));
  f16*   tdHh   = (f16*)  alloc(ALL * sizeof(f16));
  float* buC    = (float*)alloc(ALL * sizeof(float));
  float* tdCp   = (float*)alloc((size_t)NVT * BH * sizeof(float));
  float* cChild = (float*)alloc(BH * sizeof(float));
  float* yBuf   = (float*)alloc((size_t)BSZ * YLD * sizeof(float));
  f16* wEncW2 = (f16*)alloc(HD * HD * sizeof(f16));
  f16* wWiou  = (f16*)alloc(HD * 768 * sizeof(f16));
  f16* wUiou  = (f16*)alloc(HD * 768 * sizeof(f16));
  f16* wWf    = (f16*)alloc(HD * HD * sizeof(f16));
  f16* wUf    = (f16*)alloc(HD * HD * sizeof(f16));
  f16* wWih   = (f16*)alloc(HD * 1024 * sizeof(f16));
  f16* wWhh   = (f16*)alloc(HD * 1024 * sizeof(f16));
  f16* wDec1  = (f16*)alloc(512 * HD * sizeof(f16));
  (void)ws_size; (void)n_in; (void)in_sizes; (void)out_size;

  auto swz = [&](const float* s, f16* d, int K, int N) {
    swz_kernel<<<(K * N + 255) / 256, 256, 0, stream>>>(s, d, K, N);
  };
  swz(enc_w2, wEncW2, HD, HD);
  swz(W_iou, wWiou, HD, 768);
  swz(U_iou, wUiou, HD, 768);
  swz(W_f, wWf, HD, HD);
  swz(U_f, wUf, HD, HD);
  swz(W_ih, wWih, HD, 1024);
  swz(W_hh, wWhh, HD, 1024);
  swz(dec_w1, wDec1, 512, HD);

  // masked tokens feed the decoder unwritten -> zero the h mirrors
  hipMemsetAsync(buHh, 0, ALL * sizeof(f16), stream);
  hipMemsetAsync(tdHh, 0, ALL * sizeof(f16), stream);

  // Encoder
  enc1_kernel<<<NROW, 256, 0, stream>>>(cv, pv, hints, enc_w1, enc_b1, tmpH);
  gemm_af16_kernel<<<dim3(HD / 64, NROW / 64), 128, 0, stream>>>(
      tmpH, wEncW2, enc_b2, encH, HD, HD, /*relu=*/0);

  auto tok = [&](int nb, int v) -> size_t { return (size_t)(nb * NVT + v) * BH; };
  dim3 cellGrid(BSZ), cellBlk(256);

  // Bottom-up child chains (reversed scan)
  for (int k = 0; k < 2; ++k) {
    for (int v = CLEN - 1; v >= 0; --v) {
      const f16*   x  = encH + tok(k + 1, v);
      const f16*   hp = (v == CLEN - 1) ? nullptr : buHh + tok(k + 1, v + 1);
      const float* cp = (v == CLEN - 1) ? nullptr : buC + tok(k + 1, v + 1);
      step_gemm_kernel<<<dim3(16, BSZ / 64), 128, 0, stream>>>(
          x, hp, nullptr, wWiou, wUiou, wWf, wUf, yBuf, 0);
      bu_cell_kernel<<<cellGrid, cellBlk, 0, stream>>>(
          yBuf, b_iou, b_f, cp, nullptr, buHh + tok(k + 1, v), buC + tok(k + 1, v));
    }
  }

  // Bottom-up parent chain with coupling at 16 / 48
  for (int v = NVT - 1; v >= 0; --v) {
    const f16*   chHp = nullptr;
    const float* chCp = nullptr;
    if (v == 16) { chHp = buHh + tok(1, 0); chCp = buC + tok(1, 0); }
    if (v == 48) { chHp = buHh + tok(2, 0); chCp = buC + tok(2, 0); }
    int cblocks = chHp ? 20 : 16;
    const f16*   hp = (v == NVT - 1) ? nullptr : buHh + tok(0, v + 1);
    const float* cp = (v == NVT - 1) ? nullptr : buC + tok(0, v + 1);
    step_gemm_kernel<<<dim3(cblocks, BSZ / 64), 128, 0, stream>>>(
        encH + tok(0, v), hp, chHp, wWiou, wUiou, wWf, wUf, yBuf, 0);
    bu_cell_kernel<<<cellGrid, cellBlk, 0, stream>>>(
        yBuf, b_iou, b_f, cp, chCp, buHh + tok(0, v), buC + tok(0, v));
  }

  // Top-down parent chain: state starts at bottom-up root (token 0,0)
  seed_kernel<<<cellGrid, cellBlk, 0, stream>>>(
      buHh + tok(0, 0), buC + tok(0, 0), tdHh + tok(0, 0), tdCp);
  for (int v = 1; v < NVT; ++v) {
    step_gemm_kernel<<<dim3(16, BSZ / 64), 128, 0, stream>>>(
        encH + tok(0, v), tdHh + tok(0, v - 1), nullptr,
        wWih, wWhh, nullptr, nullptr, yBuf, 1);
    td_cell_kernel<<<cellGrid, cellBlk, 0, stream>>>(
        yBuf, b_ih, b_hh, tdCp + (size_t)(v - 1) * BH,
        tdHh + tok(0, v), tdCp + (size_t)v * BH);
  }

  // Top-down child chains, seeded from parent state at coupling vertex
  for (int k = 0; k < 2; ++k) {
    int ci = (k == 0) ? 16 : 48;
    for (int j = 0; j < CLEN; ++j) {
      const f16*   hp = (j == 0) ? tdHh + tok(0, ci) : tdHh + tok(k + 1, j - 1);
      const float* cp = (j == 0) ? tdCp + (size_t)ci * BH : cChild;
      step_gemm_kernel<<<dim3(16, BSZ / 64), 128, 0, stream>>>(
          encH + tok(k + 1, j), hp, nullptr, wWih, wWhh, nullptr, nullptr, yBuf, 1);
      td_cell_kernel<<<cellGrid, cellBlk, 0, stream>>>(
          yBuf, b_ih, b_hh, cp, tdHh + tok(k + 1, j), cChild);
    }
  }

  // Decoder
  gemm_dec_kernel<<<dim3(HD / 64, NROW / 64), 128, 0, stream>>>(
      buHh, tdHh, wDec1, dec_b1, tmpH);
  dec_out_kernel<<<(NROW + 255) / 256, 256, 0, stream>>>(
      tmpH, dec_w2, dec_b2, cv, pred);
}
